// saChannelBlock_53326313947743
// MI455X (gfx1250) — compile-verified, matrix operand order checked by reading, not run on hardware
//
#include <hip/hip_runtime.h>

// Problem constants (from reference): B=4, C=512, H=W=128 -> N = 16384.
#define BATCH 4
#define CC    512
#define NN    16384

typedef __attribute__((ext_vector_type(16))) __bf16 v16bf;
typedef __attribute__((ext_vector_type(8)))  __bf16 v8bf;
typedef __attribute__((ext_vector_type(8)))  float  v8f;

static __device__ __forceinline__ __bf16 f2bf(float f) {
    union { float f; unsigned u; } v; v.f = f;
    unsigned r = v.u + 0x7FFFu + ((v.u >> 16) & 1u);   // round-to-nearest-even
    unsigned short h = (unsigned short)(r >> 16);
    __bf16 o;
    __builtin_memcpy(&o, &h, sizeof(o));
    return o;
}

// ---------------------------------------------------------------------------
// Kernel 1: energy[b,i,j] = sum_n xq[b,i,n] * xk[b,j,n]   (A * B^T, NT GEMM)
// Tile 64x64, BK=64, 128 threads = 4 wave32 in a 2x2 grid, 32x32 per wave.
// Double-buffered LDS: global prefetch for tile k+1 overlaps WMMA on tile k.
// ---------------------------------------------------------------------------
#define G1_BM 64
#define G1_BN 64
#define G1_BK 64
#define G1_LDSS 72   // padded bf16 row stride (144B, 16B aligned)

__global__ __launch_bounds__(128)
void energy_gemm_kernel(const float* __restrict__ xq,
                        const float* __restrict__ xk,
                        float* __restrict__ energy) {
    __shared__ __bf16 sA[2][G1_BM * G1_LDSS];
    __shared__ __bf16 sB[2][G1_BN * G1_LDSS];

    const int b     = blockIdx.z;
    const int tileM = blockIdx.y * G1_BM;
    const int tileN = blockIdx.x * G1_BN;

    const float* Aq = xq + (size_t)b * CC * NN;
    const float* Bk = xk + (size_t)b * CC * NN;

    const int tid   = threadIdx.x;
    const int lane  = tid & 31;
    const int wave  = tid >> 5;
    const int waveM = wave >> 1;          // 0..1
    const int waveN = wave & 1;           // 0..1

    // global->LDS: 64 rows x 16 float4 per matrix
    const int lc  = tid & 15;             // float4 column
    const int lr0 = tid >> 4;             // 0..7

    v8f acc[2][2];
    #pragma unroll
    for (int i = 0; i < 2; ++i)
        #pragma unroll
        for (int j = 0; j < 2; ++j)
            acc[i][j] = (v8f)(0.0f);

    float4 ra[8], rb[8];

    auto gload = [&](int k0) {
        #pragma unroll
        for (int i = 0; i < 8; ++i) {
            const int r = lr0 + i * 8;
            ra[i] = *(const float4*)(Aq + (size_t)(tileM + r) * NN + k0 + lc * 4);
            rb[i] = *(const float4*)(Bk + (size_t)(tileN + r) * NN + k0 + lc * 4);
        }
    };
    auto lstore = [&](int bf) {
        #pragma unroll
        for (int i = 0; i < 8; ++i) {
            const int r = lr0 + i * 8;
            __bf16* pa = &sA[bf][r * G1_LDSS + lc * 4];
            pa[0] = f2bf(ra[i].x); pa[1] = f2bf(ra[i].y);
            pa[2] = f2bf(ra[i].z); pa[3] = f2bf(ra[i].w);
            __bf16* pb = &sB[bf][r * G1_LDSS + lc * 4];
            pb[0] = f2bf(rb[i].x); pb[1] = f2bf(rb[i].y);
            pb[2] = f2bf(rb[i].z); pb[3] = f2bf(rb[i].w);
        }
    };
    auto compute = [&](int bf) {
        #pragma unroll
        for (int kk = 0; kk < G1_BK; kk += 32) {
            // ISA 16-bit A/B fragment layout: lanes 0-15 take K=[0..7]+[16..23],
            // lanes 16-31 take K=[8..15]+[24..31].
            const int koff = kk + ((lane >> 4) << 3);
            const int rsel = lane & 15;
            v16bf afrag[2], bfrag[2];
            #pragma unroll
            for (int i = 0; i < 2; ++i) {
                const int row = waveM * 32 + i * 16 + rsel;
                v8bf lo = *(const v8bf*)(&sA[bf][row * G1_LDSS + koff]);
                v8bf hi = *(const v8bf*)(&sA[bf][row * G1_LDSS + koff + 16]);
                afrag[i] = __builtin_shufflevector(lo, hi, 0,1,2,3,4,5,6,7,8,9,10,11,12,13,14,15);
            }
            #pragma unroll
            for (int j = 0; j < 2; ++j) {
                const int row = waveN * 32 + j * 16 + rsel;
                v8bf lo = *(const v8bf*)(&sB[bf][row * G1_LDSS + koff]);
                v8bf hi = *(const v8bf*)(&sB[bf][row * G1_LDSS + koff + 16]);
                bfrag[j] = __builtin_shufflevector(lo, hi, 0,1,2,3,4,5,6,7,8,9,10,11,12,13,14,15);
            }
            #pragma unroll
            for (int i = 0; i < 2; ++i)
                #pragma unroll
                for (int j = 0; j < 2; ++j)
                    acc[i][j] = __builtin_amdgcn_wmma_f32_16x16x32_bf16(
                        false, afrag[i], false, bfrag[j], (short)0, acc[i][j], false, false);
        }
    };

    // Prologue: fill buffer 0.
    gload(0);
    lstore(0);
    __syncthreads();

    int buf = 0;
    for (int k0 = G1_BK; k0 < NN; k0 += G1_BK) {
        gload(k0);            // prefetch next tile (latency hidden by WMMAs)
        compute(buf);         // consume current buffer
        lstore(buf ^ 1);      // s_wait_loadcnt lands here, after compute
        __syncthreads();
        buf ^= 1;
    }
    compute(buf);             // epilogue: last tile

    // C/D layout: element r of v8f in lane L -> M = r + 8*(L>=16), N = L&15
    float* E = energy + (size_t)b * CC * CC;
    const int mh = (lane >> 4) << 3;
    const int nc = lane & 15;
    #pragma unroll
    for (int i = 0; i < 2; ++i)
        #pragma unroll
        for (int j = 0; j < 2; ++j) {
            const int mbase = tileM + waveM * 32 + i * 16 + mh;
            const int nbase = tileN + waveN * 32 + j * 16 + nc;
            #pragma unroll
            for (int r = 0; r < 8; ++r)
                E[(size_t)(mbase + r) * CC + nbase] = acc[i][j][r];
        }
}

// ---------------------------------------------------------------------------
// Kernel 2: reversed softmax in place.
// softmax(rowmax - e) == exp(rowmin_e - e_j) / sum_j exp(rowmin_e - e_j)
// One wave32 per 512-wide row, fully register resident.
// ---------------------------------------------------------------------------
__global__ __launch_bounds__(256)
void rsoftmax_kernel(float* __restrict__ energy) {
    const int wave = blockIdx.x * (blockDim.x >> 5) + (threadIdx.x >> 5);
    const int lane = threadIdx.x & 31;
    if (wave >= BATCH * CC) return;
    float* row = energy + (size_t)wave * CC;

    float v[16];
    float mn = 3.402823466e38f;
    #pragma unroll
    for (int i = 0; i < 16; ++i) {
        v[i] = row[lane + i * 32];
        mn = fminf(mn, v[i]);
    }
    #pragma unroll
    for (int off = 16; off > 0; off >>= 1)
        mn = fminf(mn, __shfl_xor(mn, off, 32));

    float s = 0.0f;
    #pragma unroll
    for (int i = 0; i < 16; ++i) {
        v[i] = __expf(mn - v[i]);
        s += v[i];
    }
    #pragma unroll
    for (int off = 16; off > 0; off >>= 1)
        s += __shfl_xor(s, off, 32);

    const float inv = 1.0f / s;
    #pragma unroll
    for (int i = 0; i < 16; ++i)
        row[lane + i * 32] = v[i] * inv;
}

// ---------------------------------------------------------------------------
// Kernel 3: out[b,i,n] = sum_j att[b,i,j] * xv[b,j,n]   (NN GEMM)
// Tile 128x128, BK=64, 256 threads = 8 wave32 in a 4x2 grid, 32x64 per wave.
// Xv tile is stored transposed into LDS so B-fragments are contiguous rows.
// Double-buffered LDS.
// ---------------------------------------------------------------------------
#define G2_BM 128
#define G2_BN 128
#define G2_BK 64
#define G2_LDSS 72

__global__ __launch_bounds__(256)
void out_gemm_kernel(const float* __restrict__ att,
                     const float* __restrict__ xv,
                     float* __restrict__ out) {
    __shared__ __bf16 sA[2][G2_BM * G2_LDSS];   // att tile: rows=m, cols=k
    __shared__ __bf16 sB[2][G2_BN * G2_LDSS];   // xv tile transposed: rows=n, cols=k

    const int b     = blockIdx.z;
    const int tileM = blockIdx.y * G2_BM;
    const int tileN = blockIdx.x * G2_BN;

    const float* A = att + (size_t)b * CC * CC;
    const float* V = xv  + (size_t)b * CC * NN;
    float*       O = out + (size_t)b * CC * NN;

    const int tid   = threadIdx.x;
    const int lane  = tid & 31;
    const int wave  = tid >> 5;
    const int waveM = wave >> 1;          // 0..3
    const int waveN = wave & 1;           // 0..1

    // A tile: 128 rows x 16 float4 cols
    const int a_lc  = tid & 15;
    const int a_lr0 = tid >> 4;           // 0..15
    // B tile: 64 k-rows x 32 float4 n-cols
    const int b_lc  = tid & 31;
    const int b_lr0 = tid >> 5;           // 0..7

    v8f acc[2][4];
    #pragma unroll
    for (int i = 0; i < 2; ++i)
        #pragma unroll
        for (int j = 0; j < 4; ++j)
            acc[i][j] = (v8f)(0.0f);

    float4 ra[8], rb[8];

    auto gload = [&](int k0) {
        #pragma unroll
        for (int i = 0; i < 8; ++i) {
            const int r = a_lr0 + i * 16;
            ra[i] = *(const float4*)(A + (size_t)(tileM + r) * CC + k0 + a_lc * 4);
        }
        #pragma unroll
        for (int i = 0; i < 8; ++i) {
            const int kr = b_lr0 + i * 8;
            rb[i] = *(const float4*)(V + (size_t)(k0 + kr) * NN + tileN + b_lc * 4);
        }
    };
    auto lstore = [&](int bf) {
        #pragma unroll
        for (int i = 0; i < 8; ++i) {
            const int r = a_lr0 + i * 16;
            __bf16* p = &sA[bf][r * G2_LDSS + a_lc * 4];
            p[0] = f2bf(ra[i].x); p[1] = f2bf(ra[i].y);
            p[2] = f2bf(ra[i].z); p[3] = f2bf(ra[i].w);
        }
        #pragma unroll
        for (int i = 0; i < 8; ++i) {
            const int kr = b_lr0 + i * 8;   // transposed store: sB[n][k]
            sB[bf][(b_lc * 4 + 0) * G2_LDSS + kr] = f2bf(rb[i].x);
            sB[bf][(b_lc * 4 + 1) * G2_LDSS + kr] = f2bf(rb[i].y);
            sB[bf][(b_lc * 4 + 2) * G2_LDSS + kr] = f2bf(rb[i].z);
            sB[bf][(b_lc * 4 + 3) * G2_LDSS + kr] = f2bf(rb[i].w);
        }
    };
    auto compute = [&](int bf) {
        #pragma unroll
        for (int kk = 0; kk < G2_BK; kk += 32) {
            const int koff = kk + ((lane >> 4) << 3);
            const int rsel = lane & 15;
            v16bf afrag[2], bfrag[4];
            #pragma unroll
            for (int i = 0; i < 2; ++i) {
                const int row = waveM * 32 + i * 16 + rsel;
                v8bf lo = *(const v8bf*)(&sA[bf][row * G2_LDSS + koff]);
                v8bf hi = *(const v8bf*)(&sA[bf][row * G2_LDSS + koff + 16]);
                afrag[i] = __builtin_shufflevector(lo, hi, 0,1,2,3,4,5,6,7,8,9,10,11,12,13,14,15);
            }
            #pragma unroll
            for (int j = 0; j < 4; ++j) {
                const int row = waveN * 64 + j * 16 + rsel;
                v8bf lo = *(const v8bf*)(&sB[bf][row * G2_LDSS + koff]);
                v8bf hi = *(const v8bf*)(&sB[bf][row * G2_LDSS + koff + 16]);
                bfrag[j] = __builtin_shufflevector(lo, hi, 0,1,2,3,4,5,6,7,8,9,10,11,12,13,14,15);
            }
            #pragma unroll
            for (int i = 0; i < 2; ++i)
                #pragma unroll
                for (int j = 0; j < 4; ++j)
                    acc[i][j] = __builtin_amdgcn_wmma_f32_16x16x32_bf16(
                        false, afrag[i], false, bfrag[j], (short)0, acc[i][j], false, false);
        }
    };

    gload(0);
    lstore(0);
    __syncthreads();

    int buf = 0;
    for (int k0 = G2_BK; k0 < CC; k0 += G2_BK) {
        gload(k0);
        compute(buf);
        lstore(buf ^ 1);
        __syncthreads();
        buf ^= 1;
    }
    compute(buf);

    const int mh = (lane >> 4) << 3;
    const int nc = lane & 15;
    #pragma unroll
    for (int i = 0; i < 2; ++i)
        #pragma unroll
        for (int j = 0; j < 4; ++j) {
            const int mbase = tileM + waveM * 32 + i * 16 + mh;
            const int nbase = tileN + waveN * 64 + j * 16 + nc;
            #pragma unroll
            for (int r = 0; r < 8; ++r)
                O[(size_t)(mbase + r) * NN + nbase] = acc[i][j][r];
        }
}

// ---------------------------------------------------------------------------
extern "C" void kernel_launch(void* const* d_in, const int* in_sizes, int n_in,
                              void* d_out, int out_size, void* d_ws, size_t ws_size,
                              hipStream_t stream) {
    (void)in_sizes; (void)n_in; (void)out_size; (void)ws_size;
    const float* xq = (const float*)d_in[0];   // x_
    const float* xk = (const float*)d_in[1];   // x_t
    const float* xv = (const float*)d_in[2];   // g_x
    float* out    = (float*)d_out;
    float* energy = (float*)d_ws;              // 4*512*512*4 = 4 MB scratch

    dim3 g1(CC / G1_BN, CC / G1_BM, BATCH);    // 8 x 8 x 4
    energy_gemm_kernel<<<g1, 128, 0, stream>>>(xq, xk, energy);

    rsoftmax_kernel<<<(BATCH * CC) / 8, 256, 0, stream>>>(energy);

    dim3 g2(NN / G2_BN, CC / G2_BM, BATCH);    // 128 x 4 x 4
    out_gemm_kernel<<<g2, 256, 0, stream>>>(energy, xv, out);
}
